// Pointnet2Backbone_44367012167881
// MI455X (gfx1250) — matrix operand order repaired and checked by reading, not compiled, not run on hardware
//
#include <hip/hip_runtime.h>
#include <hip/hip_bf16.h>

#define BATCH 8
#define NPTS  16384

typedef __attribute__((ext_vector_type(16))) _Float16 v16h;
typedef __attribute__((ext_vector_type(8)))  _Float16 v8h;
typedef __attribute__((ext_vector_type(8)))  float    v8f;

static __device__ __forceinline__ v16h cat8(v8h lo, v8h hi) {
  return __builtin_shufflevector(lo, hi, 0, 1, 2, 3, 4, 5, 6, 7,
                                 8, 9, 10, 11, 12, 13, 14, 15);
}

// ---------------------------------------------------------------------------
// Farthest point sampling: one block per batch, running min-distance array in
// LDS (<= 64KB for n=16384, fits CDNA5's 320KB WGP LDS), block argmax per step.
// Matches jnp.argmax first-index tie-break.
// ---------------------------------------------------------------------------
__global__ __launch_bounds__(256) void fps_kernel(const float* __restrict__ xyz,
                                                  int* __restrict__ out_idx,
                                                  int n, int npoint) {
  extern __shared__ float dists[];
  __shared__ float rv[256];
  __shared__ int   ri[256];
  const int b = blockIdx.x, t = threadIdx.x;
  const float* p = xyz + (size_t)b * n * 3;
  for (int j = t; j < n; j += 256) dists[j] = 1e10f;
  __syncthreads();
  int far = 0;
  for (int s = 0; s < npoint; ++s) {
    if (t == 0) out_idx[b * npoint + s] = far;
    const float cx = p[far * 3 + 0], cy = p[far * 3 + 1], cz = p[far * 3 + 2];
    float best = -1.0f; int bidx = 0x7fffffff;
    for (int j = t; j < n; j += 256) {
      const float dx = p[j * 3 + 0] - cx, dy = p[j * 3 + 1] - cy, dz = p[j * 3 + 2] - cz;
      float d = dx * dx + dy * dy + dz * dz;
      d = fminf(dists[j], d);
      dists[j] = d;
      if (d > best) { best = d; bidx = j; }   // strictly-greater keeps lowest j
    }
    rv[t] = best; ri[t] = bidx;
    __syncthreads();
    for (int off = 128; off > 0; off >>= 1) {
      if (t < off) {
        const float ov = rv[t + off]; const int oi = ri[t + off];
        if (ov > rv[t] || (ov == rv[t] && oi < ri[t])) { rv[t] = ov; ri[t] = oi; }
      }
      __syncthreads();
    }
    far = ri[0];
    __syncthreads();
  }
}

// ---------------------------------------------------------------------------
// Gather selected centroids: out[b,p,:] = xyz[b, idx[b,p], :]
// ---------------------------------------------------------------------------
__global__ void gather_xyz_kernel(const float* __restrict__ xyz, const int* __restrict__ idx,
                                  float* __restrict__ out, int n, int s, int total) {
  const int i = blockIdx.x * 256 + threadIdx.x;
  if (i >= total) return;
  const int b = i / s;
  const int j = idx[i];
  const float* src = xyz + ((size_t)b * n + j) * 3;
  float* dst = out + (size_t)i * 3;
  dst[0] = src[0]; dst[1] = src[1]; dst[2] = src[2];
}

// ---------------------------------------------------------------------------
// Ball query: first nsample in-ball indices in ascending index order (matches
// the reference's sort-based selection); tail filled with first hit (or 0).
// ---------------------------------------------------------------------------
__global__ void ball_query_kernel(const float* __restrict__ xyz, const float* __restrict__ new_xyz,
                                  int* __restrict__ idx, int n, int s, int nsample,
                                  float r2, int total) {
  const int i = blockIdx.x * 256 + threadIdx.x;
  if (i >= total) return;
  const float* q = new_xyz + (size_t)i * 3;
  const float* p = xyz + (size_t)(i / s) * n * 3;
  int* o = idx + (size_t)i * nsample;
  int cnt = 0, first = 0; bool found = false;
  for (int j = 0; j < n && cnt < nsample; ++j) {
    const float dx = p[j * 3 + 0] - q[0], dy = p[j * 3 + 1] - q[1], dz = p[j * 3 + 2] - q[2];
    if (dx * dx + dy * dy + dz * dz < r2) {
      if (!found) { found = true; first = j; }
      o[cnt++] = j;
    }
  }
  for (; cnt < nsample; ++cnt) o[cnt] = first;
}

// ---------------------------------------------------------------------------
// Grouping -> f16 rows padded to cinp (multiple of 32):
// row m = ((b*s+p)*nsample+k) -> [ (xyz[j]-center)/r , feats[j], 0-pad ]
// ---------------------------------------------------------------------------
__global__ void group_kernel(const float* __restrict__ xyz, const float* __restrict__ new_xyz,
                             const _Float16* __restrict__ feats, const int* __restrict__ idx,
                             _Float16* __restrict__ X, int n, int s, int nsample,
                             int cfeat, int cinp, float inv_r, int total) {
  const int m = blockIdx.x * 256 + threadIdx.x;
  if (m >= total) return;
  const int bp = m / nsample;          // b*s + p
  const int b  = bp / s;
  const int j  = idx[m];
  _Float16* row = X + (size_t)m * cinp;
  const float* pj = xyz + ((size_t)b * n + j) * 3;
  const float* q  = new_xyz + (size_t)bp * 3;
  row[0] = (_Float16)((pj[0] - q[0]) * inv_r);
  row[1] = (_Float16)((pj[1] - q[1]) * inv_r);
  row[2] = (_Float16)((pj[2] - q[2]) * inv_r);
  if (cfeat) {
    const _Float16* f = feats + ((size_t)b * n + j) * cfeat;
    for (int c = 0; c < cfeat; ++c) row[3 + c] = f[c];
  }
  for (int c = 3 + cfeat; c < cinp; ++c) row[c] = (_Float16)0.f;
}

// ---------------------------------------------------------------------------
// Pad-and-convert weights: W (N,K) fp32 row-major -> Wh (N,Kp) f16, zero tail.
// ---------------------------------------------------------------------------
__global__ void convert_w_kernel(const float* __restrict__ W, _Float16* __restrict__ Wh,
                                 int K, int Kp, int total) {
  const int i = blockIdx.x * 256 + threadIdx.x;
  if (i >= total) return;
  const int n = i / Kp, k = i % Kp;
  Wh[i] = (k < K) ? (_Float16)W[(size_t)n * K + k] : (_Float16)0.f;
}

// ---------------------------------------------------------------------------
// WMMA GEMM + fused epilogue: Y = relu(g*(X·Wt + b) + be)
// X: (M,KP) f16 row-major (KP mult of 32, zero-padded), Wh: (N,KP) f16.
// 128-thread block = 4 waves on 4 consecutive M tiles; each wave computes
// NREP N-tiles sharing one A fragment (X traffic / NREP, independent
// back-to-back v_wmma for the XDL pipe). Fragment layouts per
// CDNA5 ISA §7.12.2 (wave32):
//   A 16x32: lane row = lane&15; halves e0..7 -> K=k0+8h.., e8..15 -> k0+16+8h..
//   B 32x16: lane col = lane&15; same K mapping as A (e0..7 -> k0+8h.., e8..15 -> k0+16+8h..)
//   C/D:     VGPR q -> M = q + 8*(lane>>4), N = lane&15.
// All loads are 16B-aligned v8h -> global_load_b128; no masking anywhere.
// ---------------------------------------------------------------------------
template <int KP, int NREP, bool F32OUT>
__global__ __launch_bounds__(128) void wmma_gemm_kernel(
    const _Float16* __restrict__ X, const _Float16* __restrict__ Wh,
    const float* __restrict__ bias, const float* __restrict__ gain,
    const float* __restrict__ beta, void* __restrict__ Yv, int M, int N) {
  const int lane = threadIdx.x & 31;
  const int wave = threadIdx.x >> 5;
  const int h = lane >> 4, l16 = lane & 15;
  const int tileM = blockIdx.y * 4 + wave;
  const int nbase = blockIdx.x * NREP * 16;

  const _Float16* xrow = X + (size_t)(tileM * 16 + l16) * KP;
  const _Float16* wrow[NREP];
#pragma unroll
  for (int r = 0; r < NREP; ++r)
    wrow[r] = Wh + (size_t)(nbase + r * 16 + l16) * KP;

  v8f acc[NREP];
#pragma unroll
  for (int r = 0; r < NREP; ++r) acc[r] = (v8f){0.f, 0.f, 0.f, 0.f, 0.f, 0.f, 0.f, 0.f};

#pragma unroll
  for (int k0 = 0; k0 < KP; k0 += 32) {
    const v8h alo = *(const v8h*)(xrow + k0 + 8 * h);
    const v8h ahi = *(const v8h*)(xrow + k0 + 16 + 8 * h);
    const v16h a = cat8(alo, ahi);
#pragma unroll
    for (int r = 0; r < NREP; ++r) {
      const v8h blo = *(const v8h*)(wrow[r] + k0 + 8 * h);
      const v8h bhi = *(const v8h*)(wrow[r] + k0 + 16 + 8 * h);
      acc[r] = __builtin_amdgcn_wmma_f32_16x16x32_f16(false, a, false, cat8(blo, bhi),
                                                      (short)0, acc[r], false, false);
    }
  }

#pragma unroll
  for (int r = 0; r < NREP; ++r) {
    const int n = nbase + r * 16 + l16;
    const float gv = gain[n], bv = bias[n], ev = beta[n];
#pragma unroll
    for (int q = 0; q < 8; ++q) {
      const int m = tileM * 16 + q + h * 8;
      float y = gv * (acc[r][q] + bv) + ev;
      y = y > 0.f ? y : 0.f;
      if (F32OUT) ((float*)Yv)[(size_t)m * N + n] = y;
      else        ((_Float16*)Yv)[(size_t)m * N + n] = (_Float16)y;
    }
  }
}

// ---------------------------------------------------------------------------
// Max-pool over the nsample axis (f16 in / f16 out)
// ---------------------------------------------------------------------------
__global__ void maxpool_kernel(const _Float16* __restrict__ X, _Float16* __restrict__ out,
                               int nsample, int cout, int total) {
  const int i = blockIdx.x * 256 + threadIdx.x;
  if (i >= total) return;
  const int o  = i % cout;
  const int bp = i / cout;
  const _Float16* base = X + (size_t)bp * nsample * cout + o;
  float m = (float)base[0];
  for (int k = 1; k < nsample; ++k) m = fmaxf(m, (float)base[(size_t)k * cout]);
  out[(size_t)bp * cout + o] = (_Float16)m;
}

// ---------------------------------------------------------------------------
// 3-NN inverse-distance interpolation + channel concat (f16 features):
// out row = [ interp(feats2) (c2) | feats1 (c1) ]   (c1+c2 mult of 32)
// ---------------------------------------------------------------------------
__global__ void interp_concat_kernel(const float* __restrict__ xyz1, const float* __restrict__ xyz2,
                                     const _Float16* __restrict__ feats2,
                                     const _Float16* __restrict__ feats1,
                                     _Float16* __restrict__ out, int n1, int n2, int c2, int c1,
                                     int total) {
  const int i = blockIdx.x * 256 + threadIdx.x;
  if (i >= total) return;
  const int b = i / n1;
  const float* q  = xyz1 + (size_t)i * 3;
  const float* p2 = xyz2 + (size_t)b * n2 * 3;
  float d0 = 1e30f, d1 = 1e30f, d2 = 1e30f;
  int i0 = 0, i1 = 0, i2 = 0;
  for (int j = 0; j < n2; ++j) {
    const float dx = p2[j * 3 + 0] - q[0], dy = p2[j * 3 + 1] - q[1], dz = p2[j * 3 + 2] - q[2];
    const float d = dx * dx + dy * dy + dz * dz;
    if (d < d0)      { d2 = d1; i2 = i1; d1 = d0; i1 = i0; d0 = d; i0 = j; }
    else if (d < d1) { d2 = d1; i2 = i1; d1 = d;  i1 = j; }
    else if (d < d2) { d2 = d;  i2 = j; }
  }
  float w0 = 1.f / (d0 + 1e-8f), w1 = 1.f / (d1 + 1e-8f), w2 = 1.f / (d2 + 1e-8f);
  const float ws = w0 + w1 + w2;
  w0 /= ws; w1 /= ws; w2 /= ws;
  const _Float16* f0 = feats2 + ((size_t)b * n2 + i0) * c2;
  const _Float16* f1 = feats2 + ((size_t)b * n2 + i1) * c2;
  const _Float16* f2 = feats2 + ((size_t)b * n2 + i2) * c2;
  _Float16* row = out + (size_t)i * (c1 + c2);
  for (int c = 0; c < c2; ++c)
    row[c] = (_Float16)(w0 * (float)f0[c] + w1 * (float)f1[c] + w2 * (float)f2[c]);
  const _Float16* fa = feats1 + (size_t)i * c1;
  for (int c = 0; c < c1; ++c) row[c2 + c] = fa[c];
}

__global__ void copy_kernel(const float* __restrict__ src, float* __restrict__ dst, int n) {
  const int i = blockIdx.x * 256 + threadIdx.x;
  if (i < n) dst[i] = src[i];
}

// ---------------------------------------------------------------------------
// Host side
// ---------------------------------------------------------------------------
struct LayerP { const float *W, *b, *g, *be; int cin, cout; _Float16* Wh; };

#define L1D(kern, total, ...) \
  kern<<<dim3(((total) + 255) / 256), dim3(256), 0, stream>>>(__VA_ARGS__)

extern "C" void kernel_launch(void* const* d_in, const int* in_sizes, int n_in,
                              void* d_out, int out_size, void* d_ws, size_t ws_size,
                              hipStream_t stream) {
  // Network layer dims in execution order: sa1(3), sa2(2), sa3(2), sa4(2), fp1(2), fp2(2)
  static const int kCin[13]  = {3, 32, 32, 67, 64, 131, 128, 259, 256, 512, 256, 256, 128};
  static const int kCout[13] = {32, 32, 64, 64, 128, 128, 256, 256, 256, 256, 128, 128, 64};

  LayerP L[13];
  const float* pc;
  if (in_sizes[0] == BATCH * NPTS * 3) {
    // dict-insertion flattening: pointcloud, then sa1..sa4, fp1, fp2; (W,b,g,be)
    pc = (const float*)d_in[0];
    for (int i = 0; i < 13; ++i) {
      L[i].W  = (const float*)d_in[1 + 4 * i + 0];
      L[i].b  = (const float*)d_in[1 + 4 * i + 1];
      L[i].g  = (const float*)d_in[1 + 4 * i + 2];
      L[i].be = (const float*)d_in[1 + 4 * i + 3];
    }
  } else {
    // jax tree alphabetical flattening: params(fp1,fp2,sa1..sa4; W,b,be,g), pointcloud last
    static const int seq2net[13] = {9, 10, 11, 12, 0, 1, 2, 3, 4, 5, 6, 7, 8};
    for (int s = 0; s < 13; ++s) {
      const int i = seq2net[s];
      L[i].W  = (const float*)d_in[4 * s + 0];
      L[i].b  = (const float*)d_in[4 * s + 1];
      L[i].be = (const float*)d_in[4 * s + 2];
      L[i].g  = (const float*)d_in[4 * s + 3];
    }
    pc = (const float*)d_in[n_in - 1];
  }
  for (int i = 0; i < 13; ++i) { L[i].cin = kCin[i]; L[i].cout = kCout[i]; }

  // -------- workspace carving (bytes, 256B aligned), ~72 MB total --------
  char* wsb = (char*)d_ws;
  size_t off = 0;
  auto alloc = [&](size_t bytes) -> void* {
    void* p = wsb + off; off = (off + bytes + 255) & ~(size_t)255; return p;
  };
  float* xyz1   = (float*)alloc(BATCH * 512 * 3 * 4);
  float* xyz2   = (float*)alloc(BATCH * 256 * 3 * 4);
  float* xyz3   = (float*)alloc(BATCH * 64 * 3 * 4);
  float* xyz4   = (float*)alloc(BATCH * 16 * 3 * 4);
  _Float16* feats1 = (_Float16*)alloc((size_t)BATCH * 512 * 64 * 2);
  _Float16* feats2 = (_Float16*)alloc((size_t)BATCH * 256 * 128 * 2);
  _Float16* feats3 = (_Float16*)alloc((size_t)BATCH * 64 * 256 * 2);
  _Float16* feats4 = (_Float16*)alloc((size_t)BATCH * 16 * 256 * 2);
  _Float16* f1out  = (_Float16*)alloc((size_t)BATCH * 64 * 128 * 2);
  int* fpsidx  = (int*)alloc(BATCH * 512 * 4);
  int* ballidx = (int*)alloc((size_t)BATCH * 512 * 64 * 4);
  _Float16* G  = (_Float16*)alloc((size_t)262144 * 64 * 2);  // grouped / SA1-L2 out
  _Float16* A  = (_Float16*)alloc((size_t)262144 * 32 * 2);  // ping
  _Float16* Bb = (_Float16*)alloc((size_t)262144 * 32 * 2);  // pong
  for (int i = 0; i < 13; ++i) {
    const int Kp = (L[i].cin + 31) & ~31;
    L[i].Wh = (_Float16*)alloc((size_t)L[i].cout * Kp * 2);
  }
  (void)ws_size; (void)out_size;

  // weights: pad+convert fp32 -> f16 (recomputed every call: stateless)
  for (int i = 0; i < 13; ++i) {
    const int Kp = (L[i].cin + 31) & ~31;
    const int tot = L[i].cout * Kp;
    L1D(convert_w_kernel, tot, L[i].W, L[i].Wh, L[i].cin, Kp, tot);
  }

  auto gemm = [&](const _Float16* X, int li, void* Y, int M, bool f32out) {
    const LayerP& Lp = L[li];
    const int Kp = (Lp.cin + 31) & ~31;
    const int nrep = ((Lp.cout / 16) % 4 == 0) ? 4 : 2;
    dim3 grid(Lp.cout / (16 * nrep), M / 64), blk(128);
#define GEMM_CASE(KPV)                                                                        \
  case KPV:                                                                                   \
    if (f32out)        wmma_gemm_kernel<KPV, 4, true ><<<grid, blk, 0, stream>>>(X, Lp.Wh, Lp.b, Lp.g, Lp.be, Y, M, Lp.cout); \
    else if (nrep == 4) wmma_gemm_kernel<KPV, 4, false><<<grid, blk, 0, stream>>>(X, Lp.Wh, Lp.b, Lp.g, Lp.be, Y, M, Lp.cout); \
    else               wmma_gemm_kernel<KPV, 2, false><<<grid, blk, 0, stream>>>(X, Lp.Wh, Lp.b, Lp.g, Lp.be, Y, M, Lp.cout); \
    break;
    switch (Kp) {
      GEMM_CASE(32)  GEMM_CASE(64)  GEMM_CASE(96)  GEMM_CASE(128)
      GEMM_CASE(160) GEMM_CASE(256) GEMM_CASE(288) GEMM_CASE(512)
    }
#undef GEMM_CASE
  };

  // ------------------------------- SA1 -------------------------------
  fps_kernel<<<BATCH, 256, NPTS * sizeof(float), stream>>>(pc, fpsidx, NPTS, 512);
  L1D(gather_xyz_kernel, BATCH * 512, pc, fpsidx, xyz1, NPTS, 512, BATCH * 512);
  L1D(ball_query_kernel, BATCH * 512, pc, xyz1, ballidx, NPTS, 512, 64, 0.01f, BATCH * 512);
  const int M1 = BATCH * 512 * 64;
  L1D(group_kernel, M1, pc, xyz1, (const _Float16*)nullptr, ballidx, G, NPTS, 512, 64, 0, 32, 10.f, M1);
  gemm(G, 0, Bb, M1, false);
  gemm(Bb, 1, A, M1, false);
  gemm(A, 2, G, M1, false);
  L1D(maxpool_kernel, BATCH * 512 * 64, G, feats1, 64, 64, BATCH * 512 * 64);

  // ------------------------------- SA2 -------------------------------
  fps_kernel<<<BATCH, 256, 512 * sizeof(float), stream>>>(xyz1, fpsidx, 512, 256);
  L1D(gather_xyz_kernel, BATCH * 256, xyz1, fpsidx, xyz2, 512, 256, BATCH * 256);
  L1D(ball_query_kernel, BATCH * 256, xyz1, xyz2, ballidx, 512, 256, 32, 0.04f, BATCH * 256);
  const int M2 = BATCH * 256 * 32;
  L1D(group_kernel, M2, xyz1, xyz2, feats1, ballidx, G, 512, 256, 32, 64, 96, 5.f, M2);
  gemm(G, 3, A, M2, false);
  gemm(A, 4, Bb, M2, false);
  L1D(maxpool_kernel, BATCH * 256 * 128, Bb, feats2, 32, 128, BATCH * 256 * 128);

  // ------------------------------- SA3 -------------------------------
  fps_kernel<<<BATCH, 256, 256 * sizeof(float), stream>>>(xyz2, fpsidx, 256, 64);
  L1D(gather_xyz_kernel, BATCH * 64, xyz2, fpsidx, xyz3, 256, 64, BATCH * 64);
  L1D(ball_query_kernel, BATCH * 64, xyz2, xyz3, ballidx, 256, 64, 16, 0.16f, BATCH * 64);
  const int M3 = BATCH * 64 * 16;
  L1D(group_kernel, M3, xyz2, xyz3, feats2, ballidx, G, 256, 64, 16, 128, 160, 2.5f, M3);
  gemm(G, 5, A, M3, false);
  gemm(A, 6, Bb, M3, false);
  L1D(maxpool_kernel, BATCH * 64 * 256, Bb, feats3, 16, 256, BATCH * 64 * 256);

  // ------------------------------- SA4 -------------------------------
  fps_kernel<<<BATCH, 256, 64 * sizeof(float), stream>>>(xyz3, fpsidx, 64, 16);
  L1D(gather_xyz_kernel, BATCH * 16, xyz3, fpsidx, xyz4, 64, 16, BATCH * 16);
  L1D(ball_query_kernel, BATCH * 16, xyz3, xyz4, ballidx, 64, 16, 16, 0.64f, BATCH * 16);
  const int M4 = BATCH * 16 * 16;
  L1D(group_kernel, M4, xyz3, xyz4, feats3, ballidx, G, 64, 16, 16, 256, 288, 1.25f, M4);
  gemm(G, 7, A, M4, false);
  gemm(A, 8, Bb, M4, false);
  L1D(maxpool_kernel, BATCH * 16 * 256, Bb, feats4, 16, 256, BATCH * 16 * 256);

  // ------------------------------- FP1 -------------------------------
  L1D(interp_concat_kernel, BATCH * 64, xyz3, xyz4, feats4, feats3, G, 64, 16, 256, 256, BATCH * 64);
  gemm(G, 9, A, BATCH * 64, false);
  gemm(A, 10, f1out, BATCH * 64, false);

  // ------------------------------- FP2 -------------------------------
  L1D(interp_concat_kernel, BATCH * 256, xyz2, xyz3, f1out, feats2, G, 256, 64, 128, 128, BATCH * 256);
  gemm(G, 11, A, BATCH * 256, false);
  gemm(A, 12, d_out, BATCH * 256, true);   // f32 -> d_out[0 .. 131071]

  // second output: levels[1][0] = xyz after SA2
  L1D(copy_kernel, BATCH * 256 * 3, xyz2, (float*)d_out + BATCH * 256 * 64, BATCH * 256 * 3);
}
